// SRDELayer_19232863552289
// MI455X (gfx1250) — matrix operand order repaired and checked by reading, not compiled, not run on hardware
//
#include <hip/hip_runtime.h>

// ---------------- problem constants (match reference exactly) ----------------
constexpr int T_TOK = 4096;   // B*S
constexpr int HDIM  = 2048;   // H
constexpr int NEXP  = 8;      // E
constexpr int NATOM = 64;     // A
constexpr int NS    = 16384;  // sparse entries

typedef float v2f __attribute__((ext_vector_type(2)));
typedef float v8f __attribute__((ext_vector_type(8)));
typedef __bf16 bf16_t;
typedef bf16_t v8bf  __attribute__((ext_vector_type(8)));
typedef bf16_t v16bf __attribute__((ext_vector_type(16)));

// ---------------------------------------------------------------------------
// Kernel 1: gating. One wave per token; 8 expert dots share each x-load;
// wave32 shfl_xor reduction; lane 0 does clip + top-2 + softmax.
// ---------------------------------------------------------------------------
__global__ __launch_bounds__(256) void gate_kernel(
    const float* __restrict__ x, const float* __restrict__ gw,
    float* __restrict__ rw, int* __restrict__ topi) {
  const int t    = blockIdx.x * 8 + (threadIdx.x >> 5);
  const int lane = threadIdx.x & 31;

  float acc[NEXP];
#pragma unroll
  for (int e = 0; e < NEXP; ++e) acc[e] = 0.f;

  const float* xrow = x + (size_t)t * HDIM;
  for (int j = lane; j < HDIM; j += 32) {
    const float xv = xrow[j];
#pragma unroll
    for (int e = 0; e < NEXP; ++e) acc[e] += xv * gw[e * HDIM + j];
  }
#pragma unroll
  for (int e = 0; e < NEXP; ++e) {
#pragma unroll
    for (int off = 16; off > 0; off >>= 1)
      acc[e] += __shfl_xor(acc[e], off, 32);
  }
  if (lane == 0) {
    float best = -1e30f, sec = -1e30f;
    int bi = 0, si = 0;
#pragma unroll
    for (int e = 0; e < NEXP; ++e) {
      const float l = fminf(50.f, fmaxf(-50.f, acc[e]));
      if (l > best)      { sec = best; si = bi; best = l; bi = e; }
      else if (l > sec)  { sec = l; si = e; }
    }
    const float ex  = __expf(sec - best);
    const float inv = 1.f / (1.f + ex);
    rw[t * 2 + 0]   = inv;
    rw[t * 2 + 1]   = ex * inv;
    topi[t * 2 + 0] = bi;
    topi[t * 2 + 1] = si;
  }
}

// ---------------------------------------------------------------------------
// Kernel 2: weighted[e,s] = (softmax(eaw)[e,:] . atoms[:,s]) * sigmoid(imp[e,s])
// ---------------------------------------------------------------------------
__global__ __launch_bounds__(256) void weighted_kernel(
    const float* __restrict__ atoms, const float* __restrict__ eaw,
    const float* __restrict__ imp, float* __restrict__ weighted) {
  __shared__ float P[NEXP][NATOM];
  if (threadIdx.x < NEXP) {
    const int e = threadIdx.x;
    float m = -1e30f;
    for (int a = 0; a < NATOM; ++a) m = fmaxf(m, eaw[e * NATOM + a]);
    float sum = 0.f;
    for (int a = 0; a < NATOM; ++a) {
      const float v = __expf(eaw[e * NATOM + a] - m);
      P[e][a] = v;
      sum += v;
    }
    const float inv = 1.f / sum;
    for (int a = 0; a < NATOM; ++a) P[e][a] *= inv;
  }
  __syncthreads();

  const int s = blockIdx.x * 256 + threadIdx.x;
  float acc[NEXP];
#pragma unroll
  for (int e = 0; e < NEXP; ++e) acc[e] = 0.f;
  for (int a = 0; a < NATOM; ++a) {
    const float av = atoms[(size_t)a * NS + s];
#pragma unroll
    for (int e = 0; e < NEXP; ++e) acc[e] += P[e][a] * av;
  }
#pragma unroll
  for (int e = 0; e < NEXP; ++e) {
    const float im  = imp[(size_t)e * NS + s];
    const float sig = 1.f / (1.f + __expf(-im));
    weighted[(size_t)e * NS + s] = acc[e] * sig;
  }
}

// ---------------------------------------------------------------------------
// Kernel 3a: split-bf16 pre-pass: v = hi + lo, hi = bf16(v), lo = bf16(v-hi).
// ---------------------------------------------------------------------------
__global__ __launch_bounds__(256) void split_bf16_kernel(
    const float* __restrict__ src, bf16_t* __restrict__ hi,
    bf16_t* __restrict__ lo) {
  const size_t i = (size_t)blockIdx.x * 256 + threadIdx.x;
  const float v  = src[i];
  const bf16_t h = (bf16_t)v;
  hi[i] = h;
  lo[i] = (bf16_t)(v - (float)h);
}

// ---------------------------------------------------------------------------
// Kernel 3b: out = x @ W^T via split-bf16 x3: Ah*Bh + Ah*Bl + Al*Bh in f32
// accumulators using V_WMMA_F32_16X16X32_BF16 (2.7x less matrix-pipe work
// than the f32 K=4 path at the same load bytes). Wave tile 32x32, block 64x64.
// A(16-bit 16x32): lanes 0-15 K={0..7,16..23}, lanes 16-31 K={8..15,24..31}.
// B(32x16): lane = N, lanes 0-15 K={0..7,16..23}, lanes 16-31 K={8..15,24..31}.
// ---------------------------------------------------------------------------
__device__ __forceinline__ v16bf load_frag_a(const bf16_t* p) {
  const v8bf c0 = *(const v8bf*)(p);        // K chunk
  const v8bf c1 = *(const v8bf*)(p + 16);   // K chunk + 16
  v16bf r;
#pragma unroll
  for (int i = 0; i < 8; ++i) { r[i] = c0[i]; r[i + 8] = c1[i]; }
  return r;
}
__device__ __forceinline__ v16bf load_frag_b(const bf16_t* p) {
  const v8bf c0 = *(const v8bf*)(p);        // K chunk
  const v8bf c1 = *(const v8bf*)(p + 16);   // K chunk + 16
  v16bf r;
#pragma unroll
  for (int i = 0; i < 8; ++i) { r[i] = c0[i]; r[i + 8] = c1[i]; }
  return r;
}

__global__ __launch_bounds__(128) void gemm_wmma_bf16x3_kernel(
    const bf16_t* __restrict__ xh, const bf16_t* __restrict__ xl,
    const bf16_t* __restrict__ Wh, const bf16_t* __restrict__ Wl,
    float* __restrict__ out) {
  const int lane = threadIdx.x & 31;
  const int wid  = threadIdx.x >> 5;
  const int l16  = lane & 15;
  const int sel  = lane >> 4;

  const int m0 = blockIdx.y * 64 + (wid >> 1) * 32;
  const int n0 = blockIdx.x * 64 + (wid & 1) * 32;

  v8f acc00 = {}, acc01 = {}, acc10 = {}, acc11 = {};

  const size_t aRow0 = (size_t)(m0 +      l16) * HDIM + sel * 8;
  const size_t aRow1 = (size_t)(m0 + 16 + l16) * HDIM + sel * 8;
  const size_t bRow0 = (size_t)(n0 +      l16) * HDIM + sel * 8;  // B[k,n]=W[n,k]
  const size_t bRow1 = (size_t)(n0 + 16 + l16) * HDIM + sel * 8;

  for (int k0 = 0; k0 < HDIM; k0 += 32) {
    const v16bf A0h = load_frag_a(xh + aRow0 + k0);
    const v16bf A0l = load_frag_a(xl + aRow0 + k0);
    const v16bf A1h = load_frag_a(xh + aRow1 + k0);
    const v16bf A1l = load_frag_a(xl + aRow1 + k0);
    const v16bf B0h = load_frag_b(Wh + bRow0 + k0);
    const v16bf B0l = load_frag_b(Wl + bRow0 + k0);
    const v16bf B1h = load_frag_b(Wh + bRow1 + k0);
    const v16bf B1l = load_frag_b(Wl + bRow1 + k0);

    acc00 = __builtin_amdgcn_wmma_f32_16x16x32_bf16(false, A0h, false, B0h, (short)0, acc00, false, false);
    acc00 = __builtin_amdgcn_wmma_f32_16x16x32_bf16(false, A0h, false, B0l, (short)0, acc00, false, false);
    acc00 = __builtin_amdgcn_wmma_f32_16x16x32_bf16(false, A0l, false, B0h, (short)0, acc00, false, false);

    acc01 = __builtin_amdgcn_wmma_f32_16x16x32_bf16(false, A0h, false, B1h, (short)0, acc01, false, false);
    acc01 = __builtin_amdgcn_wmma_f32_16x16x32_bf16(false, A0h, false, B1l, (short)0, acc01, false, false);
    acc01 = __builtin_amdgcn_wmma_f32_16x16x32_bf16(false, A0l, false, B1h, (short)0, acc01, false, false);

    acc10 = __builtin_amdgcn_wmma_f32_16x16x32_bf16(false, A1h, false, B0h, (short)0, acc10, false, false);
    acc10 = __builtin_amdgcn_wmma_f32_16x16x32_bf16(false, A1h, false, B0l, (short)0, acc10, false, false);
    acc10 = __builtin_amdgcn_wmma_f32_16x16x32_bf16(false, A1l, false, B0h, (short)0, acc10, false, false);

    acc11 = __builtin_amdgcn_wmma_f32_16x16x32_bf16(false, A1h, false, B1h, (short)0, acc11, false, false);
    acc11 = __builtin_amdgcn_wmma_f32_16x16x32_bf16(false, A1h, false, B1l, (short)0, acc11, false, false);
    acc11 = __builtin_amdgcn_wmma_f32_16x16x32_bf16(false, A1l, false, B1h, (short)0, acc11, false, false);
  }

  const int mHi = sel * 8;
#pragma unroll
  for (int r = 0; r < 8; ++r) {
    out[(size_t)(m0 +      r + mHi) * HDIM + n0 +      l16] = acc00[r];
    out[(size_t)(m0 +      r + mHi) * HDIM + n0 + 16 + l16] = acc01[r];
    out[(size_t)(m0 + 16 + r + mHi) * HDIM + n0 +      l16] = acc10[r];
    out[(size_t)(m0 + 16 + r + mHi) * HDIM + n0 + 16 + l16] = acc11[r];
  }
}

// ---------------------------------------------------------------------------
// Kernel 3c: fallback full-fp32 GEMM via V_WMMA_F32_16X16X4_F32 (used if the
// workspace cannot hold the split-bf16 copies). Wave tile 32x32.
// ---------------------------------------------------------------------------
__global__ __launch_bounds__(128) void gemm_wmma_f32_kernel(
    const float* __restrict__ x, const float* __restrict__ W,
    float* __restrict__ out) {
  const int lane = threadIdx.x & 31;
  const int wid  = threadIdx.x >> 5;
  const int l16  = lane & 15;
  const int kOff = (lane >> 4) * 2;

  const int m0 = blockIdx.y * 64 + (wid >> 1) * 32;
  const int n0 = blockIdx.x * 64 + (wid & 1) * 32;

  v8f acc00 = {}, acc01 = {}, acc10 = {}, acc11 = {};

  const float* a0p = x + (size_t)(m0 +      l16) * HDIM + kOff;
  const float* a1p = x + (size_t)(m0 + 16 + l16) * HDIM + kOff;
  const float* b0p = W + (size_t)(n0 +      l16) * HDIM + kOff;
  const float* b1p = W + (size_t)(n0 + 16 + l16) * HDIM + kOff;

  for (int k = 0; k < HDIM; k += 4) {
    const v2f a0 = *(const v2f*)(a0p + k);
    const v2f a1 = *(const v2f*)(a1p + k);
    const v2f b0 = *(const v2f*)(b0p + k);
    const v2f b1 = *(const v2f*)(b1p + k);
    acc00 = __builtin_amdgcn_wmma_f32_16x16x4_f32(false, a0, false, b0, (short)0, acc00, false, false);
    acc01 = __builtin_amdgcn_wmma_f32_16x16x4_f32(false, a0, false, b1, (short)0, acc01, false, false);
    acc10 = __builtin_amdgcn_wmma_f32_16x16x4_f32(false, a1, false, b0, (short)0, acc10, false, false);
    acc11 = __builtin_amdgcn_wmma_f32_16x16x4_f32(false, a1, false, b1, (short)0, acc11, false, false);
  }

  const int mHi = (lane >> 4) * 8;
#pragma unroll
  for (int r = 0; r < 8; ++r) {
    out[(size_t)(m0 +      r + mHi) * HDIM + n0 +      l16] = acc00[r];
    out[(size_t)(m0 +      r + mHi) * HDIM + n0 + 16 + l16] = acc01[r];
    out[(size_t)(m0 + 16 + r + mHi) * HDIM + n0 +      l16] = acc10[r];
    out[(size_t)(m0 + 16 + r + mHi) * HDIM + n0 + 16 + l16] = acc11[r];
  }
}

// ---------------------------------------------------------------------------
// Kernel 4: sparse per-token delta via LDS ds_add_f32 accumulation.
// ---------------------------------------------------------------------------
__global__ __launch_bounds__(256) void contrib_kernel(
    const float* __restrict__ x, const float* __restrict__ weighted,
    const float* __restrict__ rw, const int* __restrict__ topi,
    const int* __restrict__ mask_idx, float* __restrict__ out) {
  __shared__ float xrow[HDIM];
  __shared__ float cacc[HDIM];
  const int t = blockIdx.x;

  for (int h = threadIdx.x; h < HDIM; h += 256) {
    xrow[h] = x[(size_t)t * HDIM + h];
    cacc[h] = 0.f;
  }
  __syncthreads();

  const float r0 = rw[t * 2 + 0];
  const float r1 = rw[t * 2 + 1];
  const float* w0 = weighted + (size_t)topi[t * 2 + 0] * NS;
  const float* w1 = weighted + (size_t)topi[t * 2 + 1] * NS;

  for (int s = threadIdx.x; s < NS; s += 256) {
    const int  mi  = mask_idx[s];
    const int  row = mi >> 11;          // / H (2048)
    const int  col = mi & (HDIM - 1);
    const float val = r0 * w0[s] + r1 * w1[s];
    atomicAdd(&cacc[row], xrow[col] * val);   // ds_add_f32
  }
  __syncthreads();

  for (int h = threadIdx.x; h < HDIM; h += 256)
    out[(size_t)t * HDIM + h] += cacc[h];
}

// ---------------------------------------------------------------------------
extern "C" void kernel_launch(void* const* d_in, const int* in_sizes, int n_in,
                              void* d_out, int out_size, void* d_ws, size_t ws_size,
                              hipStream_t stream) {
  const float* x     = (const float*)d_in[0];
  const float* gw    = (const float*)d_in[1];
  const float* W     = (const float*)d_in[2];
  const float* atoms = (const float*)d_in[3];
  const float* eaw   = (const float*)d_in[4];
  const float* imp   = (const float*)d_in[5];
  const int*   midx  = (const int*)d_in[6];
  float* out = (float*)d_out;

  // ---- workspace layout ----
  // small: rw (T*2 f32) | topi (T*2 i32) | weighted (E*NS f32)
  char* ws = (char*)d_ws;
  float* ws_rw       = (float*)ws;
  int*   ws_ti       = (int*)(ws_rw + (size_t)T_TOK * 2);
  float* ws_weighted = (float*)(ws_ti + (size_t)T_TOK * 2);
  const size_t small_bytes =
      (size_t)T_TOK * 2 * 4 + (size_t)T_TOK * 2 * 4 + (size_t)NEXP * NS * 4;

  const size_t nX = (size_t)T_TOK * HDIM;   // 8,388,608
  const size_t nW = (size_t)HDIM * HDIM;    // 4,194,304
  bf16_t* xh = (bf16_t*)(ws + small_bytes);
  bf16_t* xl = xh + nX;
  bf16_t* Wh = xl + nX;
  bf16_t* Wl = Wh + nW;
  const size_t need_bytes = small_bytes + 2 * (nX + nW) * sizeof(bf16_t);

  gate_kernel<<<T_TOK / 8, 256, 0, stream>>>(x, gw, ws_rw, ws_ti);
  weighted_kernel<<<NS / 256, 256, 0, stream>>>(atoms, eaw, imp, ws_weighted);

  if (ws_size >= need_bytes) {
    split_bf16_kernel<<<(int)(nX / 256), 256, 0, stream>>>(x, xh, xl);
    split_bf16_kernel<<<(int)(nW / 256), 256, 0, stream>>>(W, Wh, Wl);
    gemm_wmma_bf16x3_kernel<<<dim3(HDIM / 64, T_TOK / 64), 128, 0, stream>>>(
        xh, xl, Wh, Wl, out);
  } else {
    gemm_wmma_f32_kernel<<<dim3(HDIM / 64, T_TOK / 64), 128, 0, stream>>>(x, W, out);
  }

  contrib_kernel<<<T_TOK, 256, 0, stream>>>(x, ws_weighted, ws_rw, ws_ti, midx, out);
}
